// Pool_12850542149727
// MI455X (gfx1250) — 8-bit matrix operand order repaired and checked by reading, compiled, not run on hardware
//
#include <hip/hip_runtime.h>
#include <hip/hip_bf16.h>

typedef __attribute__((ext_vector_type(2))) float v2f;
typedef __attribute__((ext_vector_type(8))) float v8f;
typedef __attribute__((ext_vector_type(8))) int  v8i;

#define N_NODES 4096
#define IN_DIM  384
#define QKV_DIM 1152
#define HEAD_DIM 128
#define K_SEL   2048

#define ASYNC_CP_B128(ldsptr, gptr)                                           \
    do {                                                                      \
        unsigned _l = (unsigned)(size_t)(ldsptr);                             \
        unsigned long long _g = (unsigned long long)(size_t)(gptr);           \
        asm volatile("global_load_async_to_lds_b128 %0, %1, off"              \
                     :: "v"(_l), "v"(_g) : "memory");                         \
    } while (0)

__global__ void pack_g_kernel(const float* __restrict__ g,
                              unsigned char* __restrict__ gb,
                              unsigned char* __restrict__ gbt) {
    __shared__ unsigned char tile[32][33];
    const int bx = blockIdx.x * 32, by = blockIdx.y * 32;
    const int tx = threadIdx.x, ty = threadIdx.y;
    for (int r = 0; r < 32; r += 8) {
        float v = g[(size_t)(by + r + ty) * N_NODES + bx + tx];
        unsigned char b = (v != 0.0f) ? (unsigned char)1 : (unsigned char)0;
        gb[(size_t)(by + r + ty) * N_NODES + bx + tx] = b;
        tile[r + ty][tx] = b;
    }
    __syncthreads();
    for (int r = 0; r < 32; r += 8) {
        gbt[(size_t)(bx + r + ty) * N_NODES + by + tx] = tile[tx][r + ty];
    }
}

__global__ void qkv_wmma_kernel(const float* __restrict__ h,
                                const float* __restrict__ w,
                                const float* __restrict__ bias,
                                float* __restrict__ qkv) {
    __shared__ float As[16][IN_DIM];
    const int mtile = blockIdx.y;
    const int ncol0 = blockIdx.x * 64;
    const int t = threadIdx.x;
    for (int i = t; i < 16 * IN_DIM; i += 128) {
        int r = i / IN_DIM, c = i % IN_DIM;
        As[r][c] = h[(size_t)(mtile * 16 + r) * IN_DIM + c];
    }
    __syncthreads();
    const int wave = t >> 5, lane = t & 31;
    const int lm = lane & 15, lh = lane >> 4;
    const int ncol = ncol0 + wave * 16 + lm;
    const float* wrow = w + (size_t)ncol * IN_DIM;
    v8f acc = {};
    for (int k = 0; k < IN_DIM; k += 4) {
        v2f a, b;
        a.x = As[lm][k + 2 * lh];
        a.y = As[lm][k + 2 * lh + 1];
        b.x = wrow[k + 2 * lh];
        b.y = wrow[k + 2 * lh + 1];
        acc = __builtin_amdgcn_wmma_f32_16x16x4_f32(false, a, false, b,
                                                    (short)0, acc, false, false);
    }
    for (int r = 0; r < 8; r++) {
        int m = mtile * 16 + r + 8 * lh;
        qkv[(size_t)m * QKV_DIM + ncol] = acc[r] + bias[ncol];
    }
}

__global__ void attn_score_kernel(const float* __restrict__ qkv,
                                  const float* __restrict__ score_w,
                                  const float* __restrict__ score_b,
                                  float* __restrict__ scores) {
    __shared__ float red[128];
    __shared__ float dots[9];
    __shared__ float attn[9];
    const int n = blockIdx.x, d = threadIdx.x;
    const float* base = qkv + (size_t)n * QKV_DIM;
    float q[3], k[3], v[3];
    for (int hh = 0; hh < 3; hh++) {
        q[hh] = base[hh * HEAD_DIM + d];
        k[hh] = base[IN_DIM + hh * HEAD_DIM + d];
        v[hh] = base[2 * IN_DIM + hh * HEAD_DIM + d];
    }
    for (int p = 0; p < 9; p++) {
        red[d] = q[p / 3] * k[p % 3];
        __syncthreads();
        for (int s = 64; s > 0; s >>= 1) {
            if (d < s) red[d] += red[d + s];
            __syncthreads();
        }
        if (d == 0) dots[p] = red[0];
        __syncthreads();
    }
    if (d == 0) {
        const float scale = 0.08838834764831845f;
        for (int hh = 0; hh < 3; hh++) {
            float e0 = dots[hh * 3 + 0] * scale;
            float e1 = dots[hh * 3 + 1] * scale;
            float e2 = dots[hh * 3 + 2] * scale;
            float mx = fmaxf(e0, fmaxf(e1, e2));
            float x0 = __expf(e0 - mx), x1 = __expf(e1 - mx), x2 = __expf(e2 - mx);
            float inv = 1.0f / (x0 + x1 + x2);
            attn[hh * 3 + 0] = x0 * inv;
            attn[hh * 3 + 1] = x1 * inv;
            attn[hh * 3 + 2] = x2 * inv;
        }
    }
    __syncthreads();
    float ps = 0.0f;
    for (int hh = 0; hh < 3; hh++) {
        float o = attn[hh * 3 + 0] * v[0] + attn[hh * 3 + 1] * v[1] + attn[hh * 3 + 2] * v[2];
        ps += o * score_w[d * 3 + hh];
    }
    red[d] = ps;
    __syncthreads();
    for (int s = 64; s > 0; s >>= 1) {
        if (d < s) red[d] += red[d + s];
        __syncthreads();
    }
    if (d == 0) scores[n] = red[0] + score_b[0];
}

__global__ void rank_kernel(const float* __restrict__ scores,
                            int* __restrict__ idx, float* __restrict__ vals,
                            float* __restrict__ out_idx) {
    __shared__ int red[256];
    const int i = blockIdx.x, t = threadIdx.x;
    const float si = scores[i];
    int cnt = 0;
    for (int j = t; j < N_NODES; j += 256) {
        float sj = scores[j];
        cnt += (sj > si) || (sj == si && j < i);
    }
    red[t] = cnt;
    __syncthreads();
    for (int s = 128; s > 0; s >>= 1) {
        if (t < s) red[t] += red[t + s];
        __syncthreads();
    }
    if (t == 0) {
        int r = red[0];
        if (r < K_SEL) { idx[r] = i; vals[r] = si; out_idx[r] = (float)i; }
    }
}

__global__ void newh_kernel(const float* __restrict__ h,
                            const int* __restrict__ idx,
                            const float* __restrict__ vals,
                            float* __restrict__ out_newh) {
    const int r = blockIdx.x, t = threadIdx.x;
    const int node = idx[r];
    const float s = vals[r];
    for (int d = t; d < IN_DIM; d += 128)
        out_newh[(size_t)r * IN_DIM + d] = h[(size_t)node * IN_DIM + d] * s;
}

__global__ void __launch_bounds__(256, 1)
adj_wmma_kernel(const unsigned char* __restrict__ gb,
                const unsigned char* __restrict__ gbt,
                const int* __restrict__ idx,
                float* __restrict__ ung) {
    __shared__ int rows[128], cols[128];
    __shared__ unsigned int As[2][128][16];
    __shared__ unsigned int Bs[2][128][16];
    const int t = threadIdx.x;
    const int bx = blockIdx.x, by = blockIdx.y;
    if (t < 128) rows[t] = idx[by * 128 + t];
    else         cols[t - 128] = idx[bx * 128 + (t - 128)];
    __syncthreads();

    const int e0 = t, e1 = t + 256;
    const int ar0 = e0 >> 2, ad0 = (e0 & 3) * 4;
    const int ar1 = e1 >> 2, ad1 = (e1 & 3) * 4;
    const unsigned char* gA0 = gb  + (size_t)rows[ar0] * N_NODES + ad0 * 4;
    const unsigned char* gA1 = gb  + (size_t)rows[ar1] * N_NODES + ad1 * 4;
    const unsigned char* gB0 = gbt + (size_t)cols[ar0] * N_NODES + ad0 * 4;
    const unsigned char* gB1 = gbt + (size_t)cols[ar1] * N_NODES + ad1 * 4;

    auto issue_chunk = [&](int kk, int buf) {
        ASYNC_CP_B128(&As[buf][ar0][ad0], gA0 + kk);
        ASYNC_CP_B128(&As[buf][ar1][ad1], gA1 + kk);
        ASYNC_CP_B128(&Bs[buf][ar0][ad0], gB0 + kk);
        ASYNC_CP_B128(&Bs[buf][ar1][ad1], gB1 + kk);
    };

    issue_chunk(0, 0);
    issue_chunk(64, 1);

    const int wave = t >> 5, lane = t & 31;
    const int wm = wave >> 2, wn = wave & 3;
    const int lm = lane & 15, lh = lane >> 4;
    v8i acc[4][2] = {};

    for (int kk = 0; kk < N_NODES; kk += 64) {
        const int buf = (kk >> 6) & 1;
        if (kk + 64 < N_NODES) asm volatile("s_wait_asynccnt 0x4" ::: "memory");
        else                   asm volatile("s_wait_asynccnt 0x0" ::: "memory");
        __syncthreads();

        v8i aop[4], bop[2];
        for (int mi = 0; mi < 4; mi++) {
            int m = wm * 64 + mi * 16 + lm;
            for (int dId = 0; dId < 8; dId++) {
                int dword = 4 * (dId >> 1) + 2 * lh + (dId & 1);
                aop[mi][dId] = (int)As[buf][m][dword];
            }
        }
        for (int ni = 0; ni < 2; ni++) {
            int nn = wn * 32 + ni * 16 + lm;
            for (int vId = 0; vId < 8; vId++) {
                int dword = 4 * (vId >> 1) + 2 * lh + (vId & 1);
                bop[ni][vId] = (int)Bs[buf][nn][dword];
            }
        }
        for (int mi = 0; mi < 4; mi++)
            for (int ni = 0; ni < 2; ni++)
                acc[mi][ni] = __builtin_amdgcn_wmma_i32_16x16x64_iu8(
                    false, aop[mi], false, bop[ni], acc[mi][ni], false, false);

        __syncthreads();
        if (kk + 128 < N_NODES) issue_chunk(kk + 128, buf);
    }

    for (int mi = 0; mi < 4; mi++)
        for (int ni = 0; ni < 2; ni++)
            for (int r = 0; r < 8; r++) {
                int m = by * 128 + wm * 64 + mi * 16 + r + 8 * lh;
                int n = bx * 128 + wn * 32 + ni * 16 + lm;
                ung[(size_t)m * K_SEL + n] = (acc[mi][ni][r] != 0) ? 1.0f : 0.0f;
            }
}

extern "C" void kernel_launch(void* const* d_in, const int* in_sizes, int n_in,
                              void* d_out, int out_size, void* d_ws, size_t ws_size,
                              hipStream_t stream) {
    const float* g       = (const float*)d_in[0];
    const float* h       = (const float*)d_in[1];
    const float* qkv_w   = (const float*)d_in[3];
    const float* qkv_b   = (const float*)d_in[4];
    const float* score_w = (const float*)d_in[5];
    const float* score_b = (const float*)d_in[6];

    unsigned char* gb  = (unsigned char*)d_ws;
    unsigned char* gbt = gb + (size_t)N_NODES * N_NODES;
    float* qkv    = (float*)(gbt + (size_t)N_NODES * N_NODES);
    float* scores = qkv + (size_t)N_NODES * QKV_DIM;
    float* vals   = scores + N_NODES;
    int*   idx    = (int*)(vals + K_SEL);

    float* out_ung  = (float*)d_out;
    float* out_newh = out_ung + (size_t)K_SEL * K_SEL;
    float* out_idx  = out_newh + (size_t)K_SEL * IN_DIM;

    pack_g_kernel   <<<dim3(N_NODES / 32, N_NODES / 32), dim3(32, 8), 0, stream>>>(g, gb, gbt);
    qkv_wmma_kernel <<<dim3(QKV_DIM / 64, N_NODES / 16), 128, 0, stream>>>(h, qkv_w, qkv_b, qkv);
    attn_score_kernel<<<N_NODES, 128, 0, stream>>>(qkv, score_w, score_b, scores);
    rank_kernel     <<<N_NODES, 256, 0, stream>>>(scores, idx, vals, out_idx);
    newh_kernel     <<<K_SEL, 128, 0, stream>>>(h, idx, vals, out_newh);
    adj_wmma_kernel <<<dim3(K_SEL / 128, K_SEL / 128), 256, 0, stream>>>(gb, gbt, idx, out_ung);
}
